// CAAN_20444044329500
// MI455X (gfx1250) — compile-verified, matrix operand order checked by reading, not run on hardware
//
#include <hip/hip_runtime.h>

// ---------------------------------------------------------------------------
// MI455X (gfx1250, wave32).  B=8, N=2048, D=256.
//
// scores = softmax(q k^T / sqrt(D)) @ (v @ Ww^T) + bw
//   with  v @ Ww^T = x @ (Wv^T Ww^T) + bv.Ww     (V projection collapsed)
//
// GEMMs via v_wmma_f32_16x16x32_bf16.  Attention k-tiles staged into LDS by
// the Tensor Data Mover (tensor_load_to_lds, double-buffered, TDM row padding
// for conflict-free ds_load_b128), synced with s_wait_tensorcnt + barriers.
// ---------------------------------------------------------------------------

#define B_    8
#define N_    2048
#define D_    256
#define ROWS_ (B_ * N_)            // 16384

#define SEGS_     2                // key-dimension split per query tile
#define SEG_KEYS  (N_ / SEGS_)     // 1024
#define CHUNKS_   (SEG_KEYS / 16)  // 64 chunks of 16 keys

// LDS staging: 16 rows x 512B, TDM pads 16B after every 512B row
#define LDS_ROW_  528
#define LDS_BUF_  (16 * LDS_ROW_)  // 8448 B per buffer

typedef __attribute__((ext_vector_type(16))) __bf16 v16bf;
typedef __attribute__((ext_vector_type(8)))  __bf16 v8bf;
typedef __attribute__((ext_vector_type(8)))  float  v8f;
typedef __attribute__((ext_vector_type(4)))  unsigned int u32x4;
typedef __attribute__((ext_vector_type(8)))  int    i32x8;
typedef __attribute__((ext_vector_type(4)))  int    i32x4;

static __device__ __forceinline__ v8f wmma_bf16(v16bf a, v16bf b, v8f c) {
  return __builtin_amdgcn_wmma_f32_16x16x32_bf16(
      false, a, false, b, (short)0, c, false, false);
}

// 16x32 bf16 fragment from row-major global memory (A layout; also valid for
// the B operand when B's columns are rows of a row-major matrix).
static __device__ __forceinline__ v16bf load_frag(const __bf16* __restrict__ base,
                                                  int row, int ld, int k0, int lane) {
  int r  = lane & 15;
  int hk = (lane >> 4) << 3;                    // 0 or 8
  const __bf16* p = base + (size_t)(row + r) * ld + k0 + hk;
  v8bf lo = *(const v8bf*)(p);
  v8bf hi = *(const v8bf*)(p + 16);
  return __builtin_shufflevector(lo, hi, 0,1,2,3,4,5,6,7,8,9,10,11,12,13,14,15);
}

// ---------------------------------------------------------------------------
// TDM: DMA one 16x256 bf16 tile (row stride D_ elements) from global -> LDS,
// inserting a 4-dword pad after every 128 dwords (one 512B row) so the LDS
// row stride becomes 528B (132 dwords -> 4-bank rotation per row: the 16-lane
// stride-row fragment reads hit all 64 banks, conflict-free).
// Must be executed wave-uniformly by ONE wave per workgroup.
// clang-23 lane: 6-arg builtin (g0 u32x4, g1 i32x8, g2 i32x4, g3 i32x4,
//                               extra i32x8, cpol i32)
// ---------------------------------------------------------------------------
static __device__ __forceinline__ void tdm_load_ktile(const __bf16* gsrc,
                                                      unsigned int lds_addr) {
  unsigned long long ga = (unsigned long long)(uintptr_t)gsrc;
  u32x4 g0;
  g0[0] = 1u;                                             // count=1, user mode
  g0[1] = lds_addr;                                       // LDS byte address
  g0[2] = (unsigned int)ga;                               // global_addr[31:0]
  g0[3] = ((unsigned int)(ga >> 32) & 0x1FFFFFFu)         // global_addr[56:32]
          | 0x80000000u;                                  // type=2 ("image")
  i32x8 g1 = {
      0x07910000,      // data_size=2B | pad_enable | pad_interval=128dw | pad_amount=4dw
      0x01000000,      // tensor_dim0 = 256   (bits 79:48)
      0x08000000,      // tensor_dim1 = 2048  (bits 111:80)
      0x01000000,      // tile_dim0   = 256   (bits 127:112)
      16,              // tile_dim1   = 16
      D_,              // tensor_dim0_stride = 256 elements
      0, 0};
  i32x4 gz4 = {0, 0, 0, 0};
  i32x8 gz8 = {0, 0, 0, 0, 0, 0, 0, 0};
  __builtin_amdgcn_tensor_load_to_lds(g0, g1, gz4, gz4, gz8, 0);
}

// ---------------------------------------------------------------------------
// Kernel 0: wv_eff[d] = sum_e Wv[e,d]*Ww[e];  bias_eff = bv.Ww
// ---------------------------------------------------------------------------
__global__ void k_wveff(const float* __restrict__ Wv, const float* __restrict__ bv,
                        const float* __restrict__ Ww,
                        float* __restrict__ wv_eff, float* __restrict__ bias_eff) {
  int d = threadIdx.x;
  float acc = 0.f;
  for (int e = 0; e < D_; ++e) acc = fmaf(Wv[e * D_ + d], Ww[e], acc);
  wv_eff[d] = acc;
  if (d == 0) {
    float bb = 0.f;
    for (int e = 0; e < D_; ++e) bb = fmaf(bv[e], Ww[e], bb);
    bias_eff[0] = bb;
  }
}

// ---------------------------------------------------------------------------
// Kernel 1: Wq, Wk -> bf16
// ---------------------------------------------------------------------------
__global__ void k_convW(const float* __restrict__ Wq, const float* __restrict__ Wk,
                        __bf16* __restrict__ wqb, __bf16* __restrict__ wkb) {
  int i = blockIdx.x * blockDim.x + threadIdx.x;
  wqb[i] = (__bf16)Wq[i];
  wkb[i] = (__bf16)Wk[i];
}

// ---------------------------------------------------------------------------
// Kernel 2: x -> bf16, and sv[row] = dot(x[row,:], wv_eff) + bias_eff
// ---------------------------------------------------------------------------
__global__ void k_xconv_sv(const float* __restrict__ x,
                           const float* __restrict__ wv_eff,
                           const float* __restrict__ bias_eff,
                           __bf16* __restrict__ xb, float* __restrict__ sv) {
  __shared__ float red[256];
  int row = blockIdx.x;
  int t   = threadIdx.x;
  size_t off = (size_t)row * D_ + t;
  float v = x[off];
  xb[off] = (__bf16)v;
  red[t] = v * wv_eff[t];
  __syncthreads();
  for (int s = 128; s > 0; s >>= 1) {
    if (t < s) red[t] += red[t + s];
    __syncthreads();
  }
  if (t == 0) sv[row] = red[0] + bias_eff[0];
}

// ---------------------------------------------------------------------------
// Kernel 3: Q/K projection, one wave per 16x16 output tile.
//   qb = bf16((x@Wq^T + bq) / sqrt(D)),  kb = bf16(x@Wk^T + bk)
// ---------------------------------------------------------------------------
__global__ void k_proj_qk(const __bf16* __restrict__ xb,
                          const __bf16* __restrict__ wqb,
                          const __bf16* __restrict__ wkb,
                          const float* __restrict__ bq,
                          const float* __restrict__ bk,
                          __bf16* __restrict__ qb, __bf16* __restrict__ kb) {
  int wave = (blockIdx.x * blockDim.x + threadIdx.x) >> 5;
  int lane = threadIdx.x & 31;
  int R  = (wave >> 4) * 16;
  int E0 = (wave & 15) * 16;

  v8f cq = {}; v8f ck = {};
#pragma unroll
  for (int kk = 0; kk < 8; ++kk) {
    v16bf a    = load_frag(xb,  R,  D_, kk * 32, lane);
    v16bf bf_q = load_frag(wqb, E0, D_, kk * 32, lane);
    v16bf bf_k = load_frag(wkb, E0, D_, kk * 32, lane);
    cq = wmma_bf16(a, bf_q, cq);
    ck = wmma_bf16(a, bf_k, ck);
  }

  int   col   = E0 + (lane & 15);
  float biasq = bq[col];
  float biask = bk[col];
  const float scale = 0.0625f;          // 1/sqrt(256)
  int rbase = R + ((lane >> 4) << 3);
#pragma unroll
  for (int i = 0; i < 8; ++i) {
    size_t off = (size_t)(rbase + i) * D_ + col;
    qb[off] = (__bf16)((cq[i] + biasq) * scale);
    kb[off] = (__bf16)(ck[i] + biask);
  }
}

// ---------------------------------------------------------------------------
// Kernel 4: attention partials over one key segment.
// Block = 8 waves; wave w owns query rows [R, R+16). All waves share TDM-
// staged k chunks in LDS (double buffered). S^T = k_chunk x q^T so each lane
// owns 8 keys of one query row -> shuffle-free online softmax; halves merge
// once at the end. Writes per-row {M, L, A} partials.
// ---------------------------------------------------------------------------
__global__ void k_attn(const __bf16* __restrict__ qb, const __bf16* __restrict__ kb,
                       const float* __restrict__ sv,
                       float* __restrict__ Mp, float* __restrict__ Lp,
                       float* __restrict__ Ap) {
  __shared__ __align__(16) char smem[2 * LDS_BUF_];
  int tid  = threadIdx.x;
  int lane = tid & 31;
  int wv   = tid >> 5;                       // 0..7
  int rowgroup = blockIdx.x >> 1;            // 128 rows per block
  int seg      = blockIdx.x & 1;
  int Rg = rowgroup * 128;                   // global row base of block
  int b  = Rg / N_;
  int R  = (Rg % N_) + wv * 16;              // query row base within batch

  const __bf16* qbase = qb + (size_t)b * N_ * D_;
  const __bf16* kseg  = kb + (size_t)b * N_ * D_ + (size_t)seg * SEG_KEYS * D_;
  const float*  svb   = sv + b * N_ + seg * SEG_KEYS;
  unsigned int  lds0  = (unsigned int)(uintptr_t)&smem[0];   // LDS byte offset

  v16bf qf[8];
#pragma unroll
  for (int kk = 0; kk < 8; ++kk) qf[kk] = load_frag(qbase, R, D_, kk * 32, lane);

  if (wv == 0) tdm_load_ktile(kseg, lds0);   // prime chunk 0 -> buf 0

  float M = -3.0e38f, L = 0.f, A = 0.f;
  int hk = (lane >> 4) << 3;                 // this half's key sub-offset
  int r16 = lane & 15;

  for (int c = 0; c < CHUNKS_; ++c) {
    if (wv == 0) {
      if (c + 1 < CHUNKS_) {
        tdm_load_ktile(kseg + (size_t)(c + 1) * 16 * D_, lds0 + ((c + 1) & 1) * LDS_BUF_);
        __builtin_amdgcn_s_wait_tensorcnt(1);   // chunk c landed
      } else {
        __builtin_amdgcn_s_wait_tensorcnt(0);
      }
    }
    __syncthreads();                         // buf[c&1] ready for all waves

    const char* bufp = smem + (c & 1) * LDS_BUF_;
    v8f s = {};
#pragma unroll
    for (int kk = 0; kk < 8; ++kk) {
      int off = r16 * LDS_ROW_ + (kk * 32 + hk) * 2;
      v8bf lo = *(const v8bf*)(bufp + off);        // ds_load_b128
      v8bf hi = *(const v8bf*)(bufp + off + 32);   // ds_load_b128
      v16bf kf = __builtin_shufflevector(lo, hi,
                   0,1,2,3,4,5,6,7,8,9,10,11,12,13,14,15);
      s = wmma_bf16(kf, qf[kk], s);
    }

    // s[i] = score(key c*16+hk+i, query R+r16); scale folded into q.
    const float* svp = svb + c * 16 + hk;
    float4 sv0 = *(const float4*)(svp);
    float4 sv1 = *(const float4*)(svp + 4);
    float sva[8] = {sv0.x, sv0.y, sv0.z, sv0.w, sv1.x, sv1.y, sv1.z, sv1.w};

    float cmax = s[0];
#pragma unroll
    for (int i = 1; i < 8; ++i) cmax = fmaxf(cmax, s[i]);
    float Mn    = fmaxf(M, cmax);
    float alpha = __expf(M - Mn);
    float l = 0.f, a = 0.f;
#pragma unroll
    for (int i = 0; i < 8; ++i) {
      float p = __expf(s[i] - Mn);           // v_exp_f32, co-executes with WMMA
      l += p;
      a = fmaf(p, sva[i], a);
    }
    L = fmaf(L, alpha, l);
    A = fmaf(A, alpha, a);
    M = Mn;

    __syncthreads();                         // all waves done with buf[c&1]
  }

  // Merge the two half-wave partials (keys {0..7}+16c vs {8..15}+16c).
  float Mo = __shfl_xor(M, 16, 32);
  float Lo = __shfl_xor(L, 16, 32);
  float Ao = __shfl_xor(A, 16, 32);
  float Mn = fmaxf(M, Mo);
  float s1 = __expf(M - Mn), s2 = __expf(Mo - Mn);
  float Lt = L * s1 + Lo * s2;
  float At = A * s1 + Ao * s2;

  if (lane < 16) {
    size_t grow = (size_t)b * N_ + R + lane;
    size_t idx  = (size_t)seg * ROWS_ + grow;
    Mp[idx] = Mn;
    Lp[idx] = Lt;
    Ap[idx] = At;
  }
}

// ---------------------------------------------------------------------------
// Kernel 5: merge the 2 key-segment partials per row; add bw.
// ---------------------------------------------------------------------------
__global__ void k_merge(const float* __restrict__ Mp, const float* __restrict__ Lp,
                        const float* __restrict__ Ap, const float* __restrict__ bw,
                        float* __restrict__ out) {
  int row = blockIdx.x * blockDim.x + threadIdx.x;
  float M0 = Mp[row], M1 = Mp[ROWS_ + row];
  float L0 = Lp[row], L1 = Lp[ROWS_ + row];
  float A0 = Ap[row], A1 = Ap[ROWS_ + row];
  float Mn = fmaxf(M0, M1);
  float e0 = __expf(M0 - Mn), e1 = __expf(M1 - Mn);
  float Lt = L0 * e0 + L1 * e1;
  float At = A0 * e0 + A1 * e1;
  out[row] = At / Lt + bw[0];
}

// ---------------------------------------------------------------------------
// Launch
// ---------------------------------------------------------------------------
extern "C" void kernel_launch(void* const* d_in, const int* in_sizes, int n_in,
                              void* d_out, int out_size, void* d_ws, size_t ws_size,
                              hipStream_t stream) {
  const float* x  = (const float*)d_in[0];
  const float* Wq = (const float*)d_in[1];
  const float* bq = (const float*)d_in[2];
  const float* Wk = (const float*)d_in[3];
  const float* bk = (const float*)d_in[4];
  const float* Wv = (const float*)d_in[5];
  const float* bv = (const float*)d_in[6];
  const float* Ww = (const float*)d_in[7];
  const float* bw = (const float*)d_in[8];
  float* out = (float*)d_out;

  char* ws = (char*)d_ws;
  __bf16* qb       = (__bf16*)(ws);                 // 8 MiB
  __bf16* kb       = (__bf16*)(ws + 8388608);       // 8 MiB
  __bf16* xb       = (__bf16*)(ws + 16777216);      // 8 MiB
  __bf16* wqb      = (__bf16*)(ws + 25165824);      // 128 KiB
  __bf16* wkb      = (__bf16*)(ws + 25296896);      // 128 KiB
  float*  sv       = (float*) (ws + 25427968);      // 64 KiB
  float*  wv_eff   = (float*) (ws + 25493504);      // 1 KiB
  float*  bias_eff = (float*) (ws + 25494528);      // 256 B
  float*  Mp       = (float*) (ws + 25494784);      // 2*16384*4 = 128 KiB
  float*  Lp       = (float*) (ws + 25625856);      // 128 KiB
  float*  Ap       = (float*) (ws + 25756928);      // 128 KiB

  k_wveff   <<<1, 256, 0, stream>>>(Wv, bv, Ww, wv_eff, bias_eff);
  k_convW   <<<(D_ * D_) / 256, 256, 0, stream>>>(Wq, Wk, wqb, wkb);
  k_xconv_sv<<<ROWS_, 256, 0, stream>>>(x, wv_eff, bias_eff, xb, sv);
  k_proj_qk <<<(ROWS_ / 16) * (D_ / 16) / 8, 256, 0, stream>>>(xb, wqb, wkb, bq, bk, qb, kb);
  // 128 row-groups x 2 key segments, 8 waves/block
  k_attn    <<<(ROWS_ / 128) * SEGS_, 256, 0, stream>>>(qb, kb, sv, Mp, Lp, Ap);
  k_merge   <<<ROWS_ / 256, 256, 0, stream>>>(Mp, Lp, Ap, bw, out);
}
